// T_GRUQ_85761906966770
// MI455X (gfx1250) — compile-verified, matrix operand order checked by reading, not run on hardware
//
#include <hip/hip_runtime.h>
#include <hip/hip_bf16.h>

typedef __attribute__((ext_vector_type(16))) _Float16 v16h;
typedef __attribute__((ext_vector_type(8)))  _Float16 v8h;
typedef __attribute__((ext_vector_type(8)))  float    v8f;

#define BATCH 512
#define NEI   64
#define RREL  200
#define EDIM  128
#define SSUP  16
#define KTOP  64
#define MROWS (BATCH * KTOP)   // 32768 GRU rows per step

// ---- output section offsets (in floats), concatenated in reference return order ----
static constexpr long OUT_TREE_NODE   = 0;
static constexpr long OUT_TREE_EMB    = (long)BATCH * 3 * KTOP;                        // 98304
static constexpr long OUT_PARENT_IDX  = OUT_TREE_EMB + (long)BATCH * 3 * KTOP * EDIM;  // 12681216
static constexpr long OUT_PARENT_NODE = OUT_PARENT_IDX + (long)BATCH * 3 * KTOP;
static constexpr long OUT_AIM_REL     = OUT_PARENT_NODE + (long)BATCH * 3 * KTOP;

// ---------------------------------------------------------------------------
// order-preserving float->uint key (total order, works for negatives)
__device__ __forceinline__ unsigned int fkey(float f) {
  unsigned int u = __float_as_uint(f);
  return (u & 0x80000000u) ? ~u : (u | 0x80000000u);
}

// ---------------------------------------------------------------------------
// generic f32 -> f16 buffer conversion (one-time; weights/rel_w are L2-resident)
__global__ __launch_bounds__(256) void cvt_f16_kernel(const float* __restrict__ src,
                                                      _Float16* __restrict__ dst, int n) {
  int i = blockIdx.x * 256 + threadIdx.x;
  if (i < n) dst[i] = (_Float16)src[i];
}

// ---------------------------------------------------------------------------
// rel_score[r] = max_s cos[sup[s] * R + r]
__global__ __launch_bounds__(256) void relscore_kernel(const int* __restrict__ sup,
                                                       const float* __restrict__ cosm,
                                                       float* __restrict__ rel_score) {
  int r = threadIdx.x;
  if (r >= RREL) return;
  float m = -INFINITY;
  for (int s = 0; s < SSUP; ++s)
    m = fmaxf(m, cosm[(long)sup[s] * RREL + r]);
  rel_score[r] = m;
}

// ---------------------------------------------------------------------------
// Step 1: 64 candidates per batch, stable descending sort (== top_k of 64/64)
__global__ __launch_bounds__(64) void select1_kernel(const int* __restrict__ qh,
                                                     const int* __restrict__ edge,
                                                     const float* __restrict__ rel_score,
                                                     float* __restrict__ out,
                                                     int* __restrict__ cur,
                                                     int* __restrict__ relb) {
  __shared__ unsigned long long keys[KTOP];
  int b = blockIdx.x, t = threadIdx.x;
  int head = qh[b];
  int rel  = edge[((long)head * NEI + t) * 2 + 1];
  float sc = rel_score[rel];
  keys[t] = ((unsigned long long)fkey(sc) << 32) | (unsigned)(0xFFFFFFFFu - (unsigned)t);
  __syncthreads();
  for (int k = 2; k <= KTOP; k <<= 1)
    for (int j = k >> 1; j > 0; j >>= 1) {
      int i = t, ixj = i ^ j;
      if (ixj > i) {
        unsigned long long a = keys[i], c = keys[ixj];
        bool sw = ((i & k) == 0) ? (a < c) : (a > c);   // descending overall
        if (sw) { keys[i] = c; keys[ixj] = a; }
      }
      __syncthreads();
    }
  int idx  = (int)(0xFFFFFFFFu - (unsigned)(keys[t] & 0xFFFFFFFFull));
  int ent  = edge[((long)head * NEI + idx) * 2 + 0];
  int arel = edge[((long)head * NEI + idx) * 2 + 1];
  out[OUT_TREE_NODE   + ((long)b * 3 + 0) * KTOP + t] = (float)ent;
  out[OUT_PARENT_NODE + ((long)b * 3 + 0) * KTOP + t] = (float)head;
  cur[b * KTOP + t]  = ent;
  relb[b * KTOP + t] = arel;
}

// ---------------------------------------------------------------------------
// Steps 2/3: 4096 candidates per batch; stable top-64 via full bitonic sort in LDS
__global__ __launch_bounds__(256) void select23_kernel(const int* __restrict__ edge,
                                                       const float* __restrict__ rel_score,
                                                       const int* __restrict__ cur,
                                                       const int* __restrict__ relb_prev,
                                                       int* __restrict__ cur_next,
                                                       int* __restrict__ relb_out,
                                                       int* __restrict__ hsrc,
                                                       float* __restrict__ out,
                                                       int step /* 2 or 3 */) {
  __shared__ unsigned long long keys[NEI * KTOP];   // 4096 * 8B = 32KB (WGP LDS = 320KB)
  const int N = NEI * KTOP;
  int b = blockIdx.x, t = threadIdx.x;

  for (int c = t; c < N; c += 256) {
    int parent = c >> 6, nb = c & 63;
    int ep  = cur[b * KTOP + parent];
    int rel = edge[((long)ep * NEI + nb) * 2 + 1];
    keys[c] = ((unsigned long long)fkey(rel_score[rel]) << 32) |
              (unsigned)(0xFFFFFFFFu - (unsigned)c);
  }
  __syncthreads();
  for (int k = 2; k <= N; k <<= 1)
    for (int j = k >> 1; j > 0; j >>= 1) {
      for (int i = t; i < N; i += 256) {
        int ixj = i ^ j;
        if (ixj > i) {
          unsigned long long a = keys[i], c2 = keys[ixj];
          bool sw = ((i & k) == 0) ? (a < c2) : (a > c2);
          if (sw) { keys[i] = c2; keys[ixj] = a; }
        }
      }
      __syncthreads();
    }
  if (t < KTOP) {
    int idx    = (int)(0xFFFFFFFFu - (unsigned)(keys[t] & 0xFFFFFFFFull));
    int parent = idx >> 6, nb = idx & 63;
    int ep  = cur[b * KTOP + parent];
    int ent = edge[((long)ep * NEI + nb) * 2 + 0];
    int rel = edge[((long)ep * NEI + nb) * 2 + 1];
    int s = step - 1;   // row 1 or 2 in the (B,3,K) stacked outputs
    out[OUT_TREE_NODE   + ((long)b * 3 + s) * KTOP + t]       = (float)ent;
    out[OUT_PARENT_NODE + ((long)b * 3 + s) * KTOP + t]       = (float)ep;
    out[OUT_PARENT_IDX  + ((long)b * 3 + (s - 1)) * KTOP + t] = (float)parent;
    out[OUT_AIM_REL     + ((long)b * 3 + (s - 1)) * KTOP + t] = (float)relb_prev[b * KTOP + parent];
    if (step == 3) {
      out[OUT_PARENT_IDX + ((long)b * 3 + 2) * KTOP + t] = (float)t;
      out[OUT_AIM_REL    + ((long)b * 3 + 2) * KTOP + t] = (float)rel;
    }
    cur_next[b * KTOP + t] = ent;
    relb_out[b * KTOP + t] = rel;
    hsrc[b * KTOP + t]     = b * KTOP + parent;
  }
}

// ---------------------------------------------------------------------------
// Load one 16-half WMMA fragment from an f16 row (no conversion, pure b128 loads).
// ISA layout for 16-bit A/B 16x32 frags: per lane the 16 halfs are the two
// contiguous chunks row[k0+8*hi .. +8) and row[k0+16+8*hi .. +8).
// Caller passes p = row + k0 + 8*hi (16B aligned).
__device__ __forceinline__ v16h frag_f16(const _Float16* __restrict__ p) {
  const v8h* q = (const v8h*)p;
  v8h a = q[0], b = q[2];     // p[0..7] and p[16..23]
  v16h r;
  r[0]  = a[0]; r[1]  = a[1]; r[2]  = a[2]; r[3]  = a[3];
  r[4]  = a[4]; r[5]  = a[5]; r[6]  = a[6]; r[7]  = a[7];
  r[8]  = b[0]; r[9]  = b[1]; r[10] = b[2]; r[11] = b[3];
  r[12] = b[4]; r[13] = b[5]; r[14] = b[6]; r[15] = b[7];
  return r;
}

#define WMMA_F16(A, B, C) \
  __builtin_amdgcn_wmma_f32_16x16x32_f16(false, (A), false, (B), (short)0, (C), false, false)

// ---------------------------------------------------------------------------
// Fused GRU cell: gi = x@wih.T + bih ; gh = h@whh.T + bhh ; gates ; emb out.
// Grid: 2048 M-tiles of 16 rows. Block: 256 threads = 8 waves; wave w owns the
// 16-wide slice [16w,16w+16) of E and accumulates 6 f32 16x16 tiles (r/z/n × gi/gh).
__global__ __launch_bounds__(256) void gru_kernel(const int* __restrict__ xrel,
                                                  const int* __restrict__ hsrc,
                                                  const float* __restrict__ emb_prev,      // f32 (for z*h term)
                                                  const _Float16* __restrict__ embh_prev,  // f16 (for WMMA h operand)
                                                  const _Float16* __restrict__ relw_h,
                                                  const _Float16* __restrict__ wih_h,
                                                  const _Float16* __restrict__ whh_h,
                                                  const float* __restrict__ bih,
                                                  const float* __restrict__ bhh,
                                                  float* __restrict__ emb_out,
                                                  _Float16* __restrict__ embh_out,
                                                  float* __restrict__ out_emb,
                                                  int step) {
  const int lane = threadIdx.x & 31;
  const int wv   = threadIdx.x >> 5;        // N-tile of E: 0..7
  const int m0   = blockIdx.x * 16;
  const int r16  = lane & 15;
  const int hi   = lane >> 4;               // 0 / 1

  const int       mA   = m0 + r16;          // this lane's A-fragment row
  const _Float16* xrow = relw_h + (long)xrel[mA] * EDIM;
  const _Float16* hrow = hsrc ? (embh_prev + (long)hsrc[mA] * EDIM) : nullptr;

  const int ncol = wv * 16 + r16;           // B-fragment row in W == output column
  const _Float16* wir = wih_h + (long)(0 * EDIM + ncol) * EDIM;
  const _Float16* wiz = wih_h + (long)(1 * EDIM + ncol) * EDIM;
  const _Float16* win = wih_h + (long)(2 * EDIM + ncol) * EDIM;
  const _Float16* whr = whh_h + (long)(0 * EDIM + ncol) * EDIM;
  const _Float16* whz = whh_h + (long)(1 * EDIM + ncol) * EDIM;
  const _Float16* whn = whh_h + (long)(2 * EDIM + ncol) * EDIM;

  v8f gir = {0,0,0,0,0,0,0,0}, giz = {0,0,0,0,0,0,0,0}, gin = {0,0,0,0,0,0,0,0};
  v8f ghr = {0,0,0,0,0,0,0,0}, ghz = {0,0,0,0,0,0,0,0}, ghn = {0,0,0,0,0,0,0,0};

  const int fo = hi * 8;  // per-lane fragment chunk offset
#pragma unroll
  for (int k0 = 0; k0 < EDIM; k0 += 32) {
    v16h ax = frag_f16(xrow + k0 + fo);
    v16h br = frag_f16(wir + k0 + fo);
    v16h bz = frag_f16(wiz + k0 + fo);
    v16h bn = frag_f16(win + k0 + fo);
    gir = WMMA_F16(ax, br, gir);
    giz = WMMA_F16(ax, bz, giz);
    gin = WMMA_F16(ax, bn, gin);
    if (hrow) {                              // uniform branch: EXEC stays all-1s
      v16h ah = frag_f16(hrow + k0 + fo);
      v16h cr = frag_f16(whr + k0 + fo);
      v16h cz = frag_f16(whz + k0 + fo);
      v16h cn = frag_f16(whn + k0 + fo);
      ghr = WMMA_F16(ah, cr, ghr);
      ghz = WMMA_F16(ah, cz, ghz);
      ghn = WMMA_F16(ah, cn, ghn);
    }
  }

  const float bir = bih[ncol], biz = bih[EDIM + ncol], bin = bih[2 * EDIM + ncol];
  const float bhr = bhh[ncol], bhz = bhh[EDIM + ncol], bhn = bhh[2 * EDIM + ncol];

  // C/D layout: vgpr e, lane -> (M = m0 + e + 8*hi, N = ncol)
#pragma unroll
  for (int e = 0; e < 8; ++e) {
    int m = m0 + e + 8 * hi;
    float hv = 0.f;
    if (hsrc) hv = emb_prev[(long)hsrc[m] * EDIM + ncol];
    float gr = gir[e] + bir + ghr[e] + bhr;
    float gz = giz[e] + biz + ghz[e] + bhz;
    float r  = 1.f / (1.f + __expf(-gr));
    float z  = 1.f / (1.f + __expf(-gz));
    float hn = ghn[e] + bhn;
    float nn = tanhf(gin[e] + bin + r * hn);
    float o  = (1.f - z) * nn + z * hv;
    emb_out[(long)m * EDIM + ncol]  = o;
    embh_out[(long)m * EDIM + ncol] = (_Float16)o;
    int bb = m >> 6, kk = m & 63;
    out_emb[((long)(bb * 3 + step) * KTOP + kk) * EDIM + ncol] = o;
  }
}

// ---------------------------------------------------------------------------
extern "C" void kernel_launch(void* const* d_in, const int* in_sizes, int n_in,
                              void* d_out, int out_size, void* d_ws, size_t ws_size,
                              hipStream_t stream) {
  (void)in_sizes; (void)n_in; (void)out_size; (void)ws_size;
  const int*   query_head = (const int*)d_in[0];
  const int*   support    = (const int*)d_in[1];
  const float* cosm       = (const float*)d_in[2];
  const int*   edge       = (const int*)d_in[3];
  const float* rel_w      = (const float*)d_in[4];
  const float* wih        = (const float*)d_in[5];
  const float* whh        = (const float*)d_in[6];
  const float* bih        = (const float*)d_in[7];
  const float* bhh        = (const float*)d_in[8];
  float* out = (float*)d_out;

  char* w = (char*)d_ws;
  size_t off = 0;
  auto alloc = [&](size_t bytes) {
    void* p = w + off;
    off += (bytes + 255) & ~(size_t)255;
    return p;
  };
  float*     rel_score = (float*)alloc(RREL * sizeof(float));
  int*       curA   = (int*)alloc(MROWS * sizeof(int));
  int*       curB   = (int*)alloc(MROWS * sizeof(int));
  int*       relbA  = (int*)alloc(MROWS * sizeof(int));
  int*       relbB  = (int*)alloc(MROWS * sizeof(int));
  int*       relbC  = (int*)alloc(MROWS * sizeof(int));
  int*       hsrc2  = (int*)alloc(MROWS * sizeof(int));
  int*       hsrc3  = (int*)alloc(MROWS * sizeof(int));
  float*     emb1   = (float*)alloc((size_t)MROWS * EDIM * sizeof(float));
  float*     emb2   = (float*)alloc((size_t)MROWS * EDIM * sizeof(float));
  _Float16*  embh1  = (_Float16*)alloc((size_t)MROWS * EDIM * sizeof(_Float16));
  _Float16*  embh2  = (_Float16*)alloc((size_t)MROWS * EDIM * sizeof(_Float16));
  _Float16*  wih_h  = (_Float16*)alloc((size_t)3 * EDIM * EDIM * sizeof(_Float16));
  _Float16*  whh_h  = (_Float16*)alloc((size_t)3 * EDIM * EDIM * sizeof(_Float16));
  _Float16*  relw_h = (_Float16*)alloc((size_t)RREL * EDIM * sizeof(_Float16));

  float* out_emb = out + OUT_TREE_EMB;

  // one-time precompute: per-relation score table + f16 weight/embedding tables
  relscore_kernel<<<1, 256, 0, stream>>>(support, cosm, rel_score);
  {
    int nw = 3 * EDIM * EDIM;            // 49152
    int nr = RREL * EDIM;                // 25600
    cvt_f16_kernel<<<(nw + 255) / 256, 256, 0, stream>>>(wih, wih_h, nw);
    cvt_f16_kernel<<<(nw + 255) / 256, 256, 0, stream>>>(whh, whh_h, nw);
    cvt_f16_kernel<<<(nr + 255) / 256, 256, 0, stream>>>(rel_w, relw_h, nr);
  }

  // step 1
  select1_kernel<<<BATCH, 64, 0, stream>>>(query_head, edge, rel_score, out, curA, relbA);
  gru_kernel<<<MROWS / 16, 256, 0, stream>>>(relbA, nullptr, nullptr, nullptr,
                                             relw_h, wih_h, whh_h, bih, bhh,
                                             emb1, embh1, out_emb, 0);
  // step 2
  select23_kernel<<<BATCH, 256, 0, stream>>>(edge, rel_score, curA, relbA,
                                             curB, relbB, hsrc2, out, 2);
  gru_kernel<<<MROWS / 16, 256, 0, stream>>>(relbB, hsrc2, emb1, embh1,
                                             relw_h, wih_h, whh_h, bih, bhh,
                                             emb2, embh2, out_emb, 1);
  // step 3 (curA / emb1 / embh1 are dead by now; reuse as sinks)
  select23_kernel<<<BATCH, 256, 0, stream>>>(edge, rel_score, curB, relbB,
                                             curA, relbC, hsrc3, out, 3);
  gru_kernel<<<MROWS / 16, 256, 0, stream>>>(relbC, hsrc3, emb2, embh2,
                                             relw_h, wih_h, whh_h, bih, bhh,
                                             emb1, embh1, out_emb, 2);
}
